// TransformerBlock3D_52871047414102
// MI455X (gfx1250) — compile-verified, matrix operand order checked by reading, not run on hardware
//
#include <hip/hip_runtime.h>

// ---- problem constants (from reference) ----
#define NTOK 13824
#define C_   128
#define HEADS_ 4
#define HD_  32
#define DQ   12          // reduced grid dim (24/SR)
#define NKV  1728        // 12^3
#define CM   256         // MLP hidden
#define DSZ_ 24
#define KSR  1024        // 128 * 2*2*2 im2col K for SR conv

typedef __attribute__((ext_vector_type(16))) _Float16 v16h;
typedef __attribute__((ext_vector_type(8)))  float    v8f;

#define WMMA_F16(A_, B_, C_IN) \
  __builtin_amdgcn_wmma_f32_16x16x32_f16(false, (A_), false, (B_), (short)0, (C_IN), false, false)

// CDNA5 16-bit A/B fragment load. For A (16x32, row-major, leading dim ld):
// lane l -> row m = l&15, half = l>>4; VGPR r holds K pair
//   r<4 : k = 8*half + 2r{,+1} ; r>=4 : k = 16 + 8*half + 2(r-4){,+1}
// The same pattern loads B (32x16) from a [N][K] (column-contiguous) buffer.
__device__ __forceinline__ v16h frag_load(const _Float16* base, int ld, int lane) {
  const int mr = lane & 15;
  const int hf = lane >> 4;
  const _Float16* p = base + (size_t)mr * ld;
  v16h f;
#pragma unroll
  for (int r = 0; r < 8; ++r) {
    int k0 = (r < 4) ? (8 * hf + 2 * r) : (16 + 8 * hf + 2 * (r - 4));
    f[2 * r]     = p[k0];
    f[2 * r + 1] = p[k0 + 1];
  }
  return f;
}

// ---------------- WMMA GEMM: out[M,N] = A[M,K](f16) * Bt[N,K](f16) + bias + resid ----
// Block = 128 threads (4 waves). Block computes a 64-row x 64-col output tile:
// each wave owns 16 rows x 64 cols (4 accumulators, A-fragment reused 4x).
// The shared 64x32 B panel is staged into LDS once per k-step with
// GLOBAL_LOAD_ASYNC_TO_LDS_B128 (ASYNCcnt) and consumed by all 4 waves.
__global__ __launch_bounds__(128) void k_gemm(const _Float16* __restrict__ A,
                                              const _Float16* __restrict__ Bt,
                                              const float* __restrict__ bias,
                                              const float* __restrict__ resid,
                                              float* __restrict__ outF,
                                              _Float16* __restrict__ outH,
                                              int M, int N, int K) {
  __shared__ _Float16 Bs[64 * 32];            // 4 KB staged B panel
  const int lane = threadIdx.x & 31;
  const int wave = threadIdx.x >> 5;
  const int tn0  = blockIdx.x * 64;
  const int tm   = blockIdx.y * 4 + wave;
  const _Float16* arow = A + (size_t)tm * 16 * K;

  // staging map: 128 threads cover 64 rows x two 32-byte halves (2 x b128 each)
  const int rown  = threadIdx.x >> 1;         // 0..63 -> B row (output col)
  const int halfk = threadIdx.x & 1;          // 0/1  -> which 16-f16 half of the 32-k chunk
  const _Float16* gB = Bt + (size_t)(tn0 + rown) * K + halfk * 16;
  const unsigned  ldsoff = (unsigned)(size_t)&Bs[rown * 32 + halfk * 16];

  v8f acc[4] = {};
  for (int k0 = 0; k0 < K; k0 += 32) {
    const _Float16* src = gB + k0;
    // async copy 2 x 16B (INST_OFFSET applies to both LDS and global address)
    asm volatile("global_load_async_to_lds_b128 %0, %1, off"
                 :: "v"(ldsoff), "v"(src) : "memory");
    asm volatile("global_load_async_to_lds_b128 %0, %1, off offset:16"
                 :: "v"(ldsoff), "v"(src) : "memory");
    asm volatile("s_wait_asynccnt 0x0" ::: "memory");
    __syncthreads();

    __builtin_prefetch((const void*)(arow + (size_t)(lane & 15) * K + k0 + 64), 0, 0);
    v16h a = frag_load(arow + k0, K, lane);
#pragma unroll
    for (int j = 0; j < 4; ++j) {
      v16h b = frag_load(Bs + j * 16 * 32, 32, lane);
      acc[j] = WMMA_F16(a, b, acc[j]);
    }
    __syncthreads();
  }

  const int nc = lane & 15;
  const int m0 = tm * 16 + 8 * (lane >> 4);
#pragma unroll
  for (int j = 0; j < 4; ++j) {
    const int n = tn0 + j * 16 + nc;
    const float bv = bias ? bias[n] : 0.0f;
#pragma unroll
    for (int r = 0; r < 8; ++r) {
      size_t idx = (size_t)(m0 + r) * N + n;
      float v = acc[j][r] + bv;
      if (resid) v += resid[idx];
      if (outF) outF[idx] = v;
      if (outH) outH[idx] = (_Float16)v;
    }
  }
}

// ---------------- LayerNorm (one wave per row) ----------------
__global__ __launch_bounds__(256) void k_ln(const float* __restrict__ x,
                                            const float* __restrict__ g,
                                            const float* __restrict__ b,
                                            float* __restrict__ outF,
                                            _Float16* __restrict__ outH,
                                            int R, int W) {
  int row  = blockIdx.x * 8 + (threadIdx.x >> 5);
  int lane = threadIdx.x & 31;
  if (row >= R) return;
  const float* xr = x + (size_t)row * W;
  float s = 0.0f, s2 = 0.0f;
  for (int c = lane; c < W; c += 32) { float v = xr[c]; s += v; s2 += v * v; }
#pragma unroll
  for (int m = 16; m >= 1; m >>= 1) { s += __shfl_xor(s, m, 32); s2 += __shfl_xor(s2, m, 32); }
  float mean = s / W;
  float rs = rsqrtf(s2 / W - mean * mean + 1e-5f);
  for (int c = lane; c < W; c += 32) {
    float v = (xr[c] - mean) * rs * g[c] + b[c];
    if (outF) outF[(size_t)row * W + c] = v;
    if (outH) outH[(size_t)row * W + c] = (_Float16)v;
  }
}

// ---------------- weight transpose f32[K,N] -> f16[N,K] ----------------
__global__ void k_wt(const float* __restrict__ w, _Float16* __restrict__ wt, int K, int N) {
  int i = blockIdx.x * blockDim.x + threadIdx.x;
  if (i >= K * N) return;
  int k = i / N, n = i % N;
  wt[(size_t)n * K + k] = (_Float16)w[i];
}

// ---------------- flat f32 -> f16 cast ----------------
__global__ void k_cast(const float* __restrict__ w, _Float16* __restrict__ o, int n) {
  int i = blockIdx.x * blockDim.x + threadIdx.x;
  if (i < n) o[i] = (_Float16)w[i];
}

// ---------------- im2col pack for stride-2 2x2x2 SR conv ----------------
__global__ void k_srpack(const float* __restrict__ h, _Float16* __restrict__ A) {
  int idx = blockIdx.x * blockDim.x + threadIdx.x;
  if (idx >= NKV * KSR) return;
  int j = idx % KSR, t = idx / KSR;
  int kx = j & 1, ky = (j >> 1) & 1, kz = (j >> 2) & 1, i = j >> 3;
  int wx = t % DQ, hy = (t / DQ) % DQ, dz = t / (DQ * DQ);
  int vox = ((2 * dz + kz) * DSZ_ + (2 * hy + ky)) * DSZ_ + (2 * wx + kx);
  A[idx] = (_Float16)h[(size_t)vox * C_ + i];
}

// ---------------- offset 3x3x3 conv (12 out ch) + tanh * 2 ----------------
__global__ void k_offconv(const float* __restrict__ xst, const float* __restrict__ offw,
                          const float* __restrict__ offb, float* __restrict__ offc) {
  int idx = blockIdx.x * blockDim.x + threadIdx.x;
  if (idx >= 12 * NKV) return;
  int oc = idx / NKV, m = idx % NKV;
  int wx = m % DQ, hy = (m / DQ) % DQ, dz = m / (DQ * DQ);
  float acc = offb[oc];
  for (int kz = 0; kz < 3; ++kz) {
    int z = dz + kz - 1; if (z < 0 || z >= DQ) continue;
    for (int ky = 0; ky < 3; ++ky) {
      int y = hy + ky - 1; if (y < 0 || y >= DQ) continue;
      for (int kx = 0; kx < 3; ++kx) {
        int x = wx + kx - 1; if (x < 0 || x >= DQ) continue;
        int nm  = (z * DQ + y) * DQ + x;
        int tap = (kz * 3 + ky) * 3 + kx;
        const float* xr = xst + (size_t)nm * C_;
        const float* wr = offw + (size_t)oc * C_ * 27 + tap;
        float s = 0.0f;
        for (int c = 0; c < C_; ++c) s += xr[c] * wr[(size_t)c * 27];
        acc += s;
      }
    }
  }
  offc[idx] = tanhf(acc) * 2.0f;  // OFR
}

// ---------------- trilinear grid-sample of k,v (reference z/x channel quirk kept) ----------------
__global__ __launch_bounds__(32) void k_gsample(const float* __restrict__ kv,
                                                const float* __restrict__ offc,
                                                _Float16* __restrict__ sk16,
                                                _Float16* __restrict__ svt16) {
  const int head = blockIdx.y;
  const int m    = blockIdx.x;
  const int ch   = threadIdx.x;   // 0..31
  int wx = m % DQ, hy = (m / DQ) % DQ, dz = m / (DQ * DQ);
  const float step = 2.0f / (DQ - 1);
  // samp[...,0]=bz+off0 used as x, [...,1]=by+off1 as y, [...,2]=bx+off2 as z
  float u0 = -1.0f + dz * step + offc[(head * 3 + 0) * NKV + m];
  float u1 = -1.0f + hy * step + offc[(head * 3 + 1) * NKV + m];
  float u2 = -1.0f + wx * step + offc[(head * 3 + 2) * NKV + m];
  float gx = (u0 + 1.0f) * 0.5f * (DQ - 1);
  float gy = (u1 + 1.0f) * 0.5f * (DQ - 1);
  float gz = (u2 + 1.0f) * 0.5f * (DQ - 1);
  float fx0 = floorf(gx), fy0 = floorf(gy), fz0 = floorf(gz);
  float fx = gx - fx0, fy = gy - fy0, fz = gz - fz0;
  int x0 = (int)fx0, y0 = (int)fy0, z0 = (int)fz0;
  float ak = 0.0f, av = 0.0f;
#pragma unroll
  for (int di = 0; di < 2; ++di) {
    int zi = z0 + di; float wz = di ? fz : 1.0f - fz;
#pragma unroll
    for (int dj = 0; dj < 2; ++dj) {
      int yi = y0 + dj; float wy = dj ? fy : 1.0f - fy;
#pragma unroll
      for (int dk = 0; dk < 2; ++dk) {
        int xi = x0 + dk; float ww = dk ? fx : 1.0f - fx;
        if (zi < 0 || zi >= DQ || yi < 0 || yi >= DQ || xi < 0 || xi >= DQ) continue;
        int nm = (zi * DQ + yi) * DQ + xi;
        float w3 = wz * wy * ww;
        ak += w3 * kv[(size_t)nm * (2 * C_) + head * HD_ + ch];
        av += w3 * kv[(size_t)nm * (2 * C_) + C_ + head * HD_ + ch];
      }
    }
  }
  sk16[((size_t)head * NKV + m) * HD_ + ch] = (_Float16)ak;           // [h][m][d]
  svt16[((size_t)head * HD_ + ch) * NKV + m] = (_Float16)av;          // [h][d][m]
}

// ---------------- flash attention: wave32, 16 queries/wave, d=32=K ----------------
__global__ __launch_bounds__(128) void k_attn(const _Float16* __restrict__ q16,
                                              const _Float16* __restrict__ sk16,
                                              const _Float16* __restrict__ svt16,
                                              _Float16* __restrict__ o16) {
  __shared__ _Float16 lsP[4][16 * 32];   // per-wave P tile for C-frag -> A-frag relayout
  const int lane = threadIdx.x & 31;
  const int wave = threadIdx.x >> 5;
  const int head = blockIdx.y;
  const int tm   = blockIdx.x * 4 + wave;      // 0..863
  _Float16* Pw = lsP[wave];

  v16h aq = frag_load(q16 + (size_t)tm * 16 * C_ + head * HD_, C_, lane);

  float mi[8], li[8];
  v8f acc0 = {}, acc1 = {};
#pragma unroll
  for (int r = 0; r < 8; ++r) { mi[r] = -1e30f; li[r] = 0.0f; }
  const float scale = 0.17677669529663689f;   // 32^-0.5
  const _Float16* skh = sk16  + (size_t)head * NKV * HD_;
  const _Float16* svh = svt16 + (size_t)head * HD_ * NKV;

  const int col   = lane & 15;
  const int rbase = 8 * (lane >> 4);

  for (int c0 = 0; c0 < NKV; c0 += 32) {
    // S = q (16x32) @ skT : two 16x16 tiles
    v16h bk0 = frag_load(skh + (size_t)c0 * HD_, HD_, lane);
    v16h bk1 = frag_load(skh + (size_t)(c0 + 16) * HD_, HD_, lane);
    v8f z0 = {}, z1 = {};
    v8f s0 = WMMA_F16(aq, bk0, z0);
    v8f s1 = WMMA_F16(aq, bk1, z1);
    float p0[8], p1[8];
#pragma unroll
    for (int r = 0; r < 8; ++r) {
      float a = s0[r] * scale, b = s1[r] * scale;
      float mx = fmaxf(a, b);
      mx = fmaxf(mx, __shfl_xor(mx, 1, 32));
      mx = fmaxf(mx, __shfl_xor(mx, 2, 32));
      mx = fmaxf(mx, __shfl_xor(mx, 4, 32));
      mx = fmaxf(mx, __shfl_xor(mx, 8, 32));
      float mnew  = fmaxf(mi[r], mx);
      float alpha = __expf(mi[r] - mnew);
      float e0 = __expf(a - mnew);
      float e1 = __expf(b - mnew);
      float rs = e0 + e1;
      rs += __shfl_xor(rs, 1, 32);
      rs += __shfl_xor(rs, 2, 32);
      rs += __shfl_xor(rs, 4, 32);
      rs += __shfl_xor(rs, 8, 32);
      li[r] = li[r] * alpha + rs;
      mi[r] = mnew;
      acc0[r] *= alpha;
      acc1[r] *= alpha;
      p0[r] = e0; p1[r] = e1;
    }
    // write P (16 rows x 32 keys) to this wave's private LDS tile, re-read in A-frag layout.
    // DS ops from one wave stay in order (DScnt), so no cross-wave barrier needed.
#pragma unroll
    for (int r = 0; r < 8; ++r) {
      Pw[(rbase + r) * 32 + col]      = (_Float16)p0[r];
      Pw[(rbase + r) * 32 + 16 + col] = (_Float16)p1[r];
    }
    v16h ap  = frag_load(Pw, 32, lane);
    // O += P (16x32keys) @ V (32keys x 32d) : two 16x16 output tiles
    v16h bv0 = frag_load(svh + c0, NKV, lane);
    v16h bv1 = frag_load(svh + (size_t)16 * NKV + c0, NKV, lane);
    acc0 = WMMA_F16(ap, bv0, acc0);
    acc1 = WMMA_F16(ap, bv1, acc1);
  }

  const int m0 = tm * 16 + rbase;
#pragma unroll
  for (int r = 0; r < 8; ++r) {
    float inv = 1.0f / fmaxf(li[r], 1e-20f);
    o16[(size_t)(m0 + r) * C_ + head * HD_ + col]      = (_Float16)(acc0[r] * inv);
    o16[(size_t)(m0 + r) * C_ + head * HD_ + 16 + col] = (_Float16)(acc1[r] * inv);
  }
}

// ---------------- depthwise 3x3x3 conv ----------------
__global__ void k_dw(const float* __restrict__ y, const float* __restrict__ dww,
                     const float* __restrict__ dwb, float* __restrict__ yc) {
  int idx = blockIdx.x * blockDim.x + threadIdx.x;
  if (idx >= NTOK * CM) return;
  int c = idx % CM, t = idx / CM;
  int x = t % DSZ_, yy = (t / DSZ_) % DSZ_, z = t / (DSZ_ * DSZ_);
  float acc = dwb[c];
  for (int kz = 0; kz < 3; ++kz) {
    int zz = z + kz - 1; if (zz < 0 || zz >= DSZ_) continue;
    for (int ky = 0; ky < 3; ++ky) {
      int yz = yy + ky - 1; if (yz < 0 || yz >= DSZ_) continue;
      for (int kx = 0; kx < 3; ++kx) {
        int xz = x + kx - 1; if (xz < 0 || xz >= DSZ_) continue;
        int nt = (zz * DSZ_ + yz) * DSZ_ + xz;
        acc += y[(size_t)nt * CM + c] * dww[c * 27 + (kz * 3 + ky) * 3 + kx];
      }
    }
  }
  yc[idx] = acc;
}

// ---------------- BN batch stats (block per channel) ----------------
__global__ __launch_bounds__(256) void k_bnstat(const float* __restrict__ yc,
                                                float* __restrict__ mu,
                                                float* __restrict__ rvar) {
  __shared__ float s1[256], s2[256];
  int c = blockIdx.x, tid = threadIdx.x;
  float a = 0.0f, b = 0.0f;
  for (int t = tid; t < NTOK; t += 256) {
    float v = yc[(size_t)t * CM + c];
    a += v; b += v * v;
  }
  s1[tid] = a; s2[tid] = b; __syncthreads();
  for (int st = 128; st > 0; st >>= 1) {
    if (tid < st) { s1[tid] += s1[tid + st]; s2[tid] += s2[tid + st]; }
    __syncthreads();
  }
  if (tid == 0) {
    float mean = s1[0] / NTOK;
    float var  = s2[0] / NTOK - mean * mean;
    mu[c] = mean; rvar[c] = rsqrtf(var + 1e-5f);
  }
}

// ---------------- BN apply + exact GELU -> f16 ----------------
__global__ void k_bnact(const float* __restrict__ yc, const float* __restrict__ mu,
                        const float* __restrict__ rvar, const float* __restrict__ g,
                        const float* __restrict__ b, _Float16* __restrict__ out16) {
  int idx = blockIdx.x * blockDim.x + threadIdx.x;
  if (idx >= NTOK * CM) return;
  int c = idx % CM;
  float v  = (yc[idx] - mu[c]) * rvar[c] * g[c] + b[c];
  float ge = 0.5f * v * (1.0f + erff(v * 0.70710678118654752f));
  out16[idx] = (_Float16)ge;
}

extern "C" void kernel_launch(void* const* d_in, const int* in_sizes, int n_in,
                              void* d_out, int out_size, void* d_ws, size_t ws_size,
                              hipStream_t stream) {
  (void)in_sizes; (void)n_in; (void)out_size; (void)ws_size;
  const float* x    = (const float*)d_in[0];
  const float* n1g  = (const float*)d_in[1];
  const float* n1b  = (const float*)d_in[2];
  const float* qw   = (const float*)d_in[3];
  const float* kvw  = (const float*)d_in[4];
  const float* pw   = (const float*)d_in[5];
  const float* pb   = (const float*)d_in[6];
  const float* srw  = (const float*)d_in[7];
  const float* srb  = (const float*)d_in[8];
  const float* sng  = (const float*)d_in[9];
  const float* snb  = (const float*)d_in[10];
  const float* ow   = (const float*)d_in[11];
  const float* ob   = (const float*)d_in[12];
  const float* n2g  = (const float*)d_in[13];
  const float* n2b  = (const float*)d_in[14];
  const float* f1w  = (const float*)d_in[15];
  const float* f1b  = (const float*)d_in[16];
  const float* dww  = (const float*)d_in[17];
  const float* dwb  = (const float*)d_in[18];
  const float* bng  = (const float*)d_in[19];
  const float* bnb  = (const float*)d_in[20];
  const float* f2w  = (const float*)d_in[21];
  const float* f2b  = (const float*)d_in[22];

  size_t off = 0;
  char* base = (char*)d_ws;
  auto alloc = [&](size_t bytes) -> void* {
    void* p = base + off;
    off += (bytes + 255) & ~(size_t)255;
    return p;
  };

  float*     h      = (float*)    alloc((size_t)NTOK * C_ * 4);
  _Float16*  h16    = (_Float16*) alloc((size_t)NTOK * C_ * 2);
  _Float16*  q16    = (_Float16*) alloc((size_t)NTOK * C_ * 2);
  _Float16*  qwT    = (_Float16*) alloc((size_t)C_ * C_ * 2);
  _Float16*  kvwT   = (_Float16*) alloc((size_t)C_ * 2 * C_ * 2);
  _Float16*  pwT    = (_Float16*) alloc((size_t)C_ * C_ * 2);
  _Float16*  f1wT   = (_Float16*) alloc((size_t)C_ * CM * 2);
  _Float16*  f2wT   = (_Float16*) alloc((size_t)CM * C_ * 2);
  _Float16*  srw16  = (_Float16*) alloc((size_t)C_ * KSR * 2);
  _Float16*  Asr    = (_Float16*) alloc((size_t)NKV * KSR * 2);
  float*     xsconv = (float*)    alloc((size_t)NKV * C_ * 4);
  float*     xst    = (float*)    alloc((size_t)NKV * C_ * 4);
  _Float16*  xst16  = (_Float16*) alloc((size_t)NKV * C_ * 2);
  float*     kvbuf  = (float*)    alloc((size_t)NKV * 2 * C_ * 4);
  float*     offc   = (float*)    alloc((size_t)12 * NKV * 4);
  _Float16*  sk16   = (_Float16*) alloc((size_t)HEADS_ * NKV * HD_ * 2);
  _Float16*  svt16  = (_Float16*) alloc((size_t)HEADS_ * HD_ * NKV * 2);
  _Float16*  o16    = (_Float16*) alloc((size_t)NTOK * C_ * 2);
  float*     x1     = (float*)    alloc((size_t)NTOK * C_ * 4);
  _Float16*  h216   = (_Float16*) alloc((size_t)NTOK * C_ * 2);
  float*     ybuf   = (float*)    alloc((size_t)NTOK * CM * 4);
  float*     ycbuf  = (float*)    alloc((size_t)NTOK * CM * 4);
  float*     mu     = (float*)    alloc((size_t)CM * 4);
  float*     rvar   = (float*)    alloc((size_t)CM * 4);
  _Float16*  g16    = (_Float16*) alloc((size_t)NTOK * CM * 2);

  // ---- weight conversions ----
  k_wt  <<<dim3((C_ * C_ + 255) / 256),       256, 0, stream>>>(qw,  qwT,  C_, C_);
  k_wt  <<<dim3((C_ * 2 * C_ + 255) / 256),   256, 0, stream>>>(kvw, kvwT, C_, 2 * C_);
  k_wt  <<<dim3((C_ * C_ + 255) / 256),       256, 0, stream>>>(pw,  pwT,  C_, C_);
  k_wt  <<<dim3((C_ * CM + 255) / 256),       256, 0, stream>>>(f1w, f1wT, C_, CM);
  k_wt  <<<dim3((CM * C_ + 255) / 256),       256, 0, stream>>>(f2w, f2wT, CM, C_);
  k_cast<<<dim3((C_ * KSR + 255) / 256),      256, 0, stream>>>(srw, srw16, C_ * KSR); // already [O][I*2*2*2]=[N][K]

  // ---- norm1 -> h ----
  k_ln<<<dim3(NTOK / 8), 256, 0, stream>>>(x, n1g, n1b, h, h16, NTOK, C_);

  // ---- q = h @ qw ----
  k_gemm<<<dim3(C_ / 64, (NTOK / 16) / 4), 128, 0, stream>>>(h16, qwT, nullptr, nullptr,
                                                             nullptr, q16, NTOK, C_, C_);

  // ---- SR conv as im2col GEMM, then sr_norm ----
  k_srpack<<<dim3((NKV * KSR) / 256), 256, 0, stream>>>(h, Asr);
  k_gemm<<<dim3(C_ / 64, (NKV / 16) / 4), 128, 0, stream>>>(Asr, srw16, srb, nullptr,
                                                            xsconv, nullptr, NKV, C_, KSR);
  k_ln<<<dim3(NKV / 8), 256, 0, stream>>>(xsconv, sng, snb, xst, xst16, NKV, C_);

  // ---- kv = xs_t @ kv_w ----
  k_gemm<<<dim3((2 * C_) / 64, (NKV / 16) / 4), 128, 0, stream>>>(xst16, kvwT, nullptr, nullptr,
                                                                  kvbuf, nullptr, NKV, 2 * C_, C_);

  // ---- offsets + grid sample ----
  k_offconv<<<dim3((12 * NKV + 127) / 128), 128, 0, stream>>>(xst, ow, ob, offc);
  k_gsample<<<dim3(NKV, HEADS_), 32, 0, stream>>>(kvbuf, offc, sk16, svt16);

  // ---- flash attention ----
  k_attn<<<dim3((NTOK / 16) / 4, HEADS_), 128, 0, stream>>>(q16, sk16, svt16, o16);

  // ---- proj + residual -> x1 ----
  k_gemm<<<dim3(C_ / 64, (NTOK / 16) / 4), 128, 0, stream>>>(o16, pwT, pb, x,
                                                             x1, nullptr, NTOK, C_, C_);

  // ---- MLP branch ----
  k_ln<<<dim3(NTOK / 8), 256, 0, stream>>>(x1, n2g, n2b, nullptr, h216, NTOK, C_);
  k_gemm<<<dim3(CM / 64, (NTOK / 16) / 4), 128, 0, stream>>>(h216, f1wT, f1b, nullptr,
                                                             ybuf, nullptr, NTOK, CM, C_);
  k_dw<<<dim3((NTOK * CM) / 256), 256, 0, stream>>>(ybuf, dww, dwb, ycbuf);
  k_bnstat<<<dim3(CM), 256, 0, stream>>>(ycbuf, mu, rvar);
  k_bnact<<<dim3((NTOK * CM) / 256), 256, 0, stream>>>(ycbuf, mu, rvar, bng, bnb, g16);
  k_gemm<<<dim3(C_ / 64, (NTOK / 16) / 4), 128, 0, stream>>>(g16, f2wT, f2b, x1,
                                                             (float*)d_out, nullptr, NTOK, C_, CM);
}
